// Attention_RPEHP_76613626626348
// MI455X (gfx1250) — compile-verified, matrix operand order checked by reading, not run on hardware
//
#include <hip/hip_runtime.h>

// ---------------- problem constants ----------------
#define H_    32
#define W_    32
#define N_TOK 1024       // H_*W_
#define HEADS 8
#define DH    64
#define DIM   512
#define INNER 512        // HEADS*DH
#define B_    8
#define NDELTA 3969      // 63*63 relative positions

typedef __attribute__((ext_vector_type(16))) __bf16 v16bf;
typedef __attribute__((ext_vector_type(8)))  float  v8f;

#if defined(__has_builtin)
#if __has_builtin(__builtin_amdgcn_tensor_load_to_lds)
#define HAVE_TDM 1
#endif
#endif
#ifndef HAVE_TDM
#define HAVE_TDM 0
#endif

union FragU {
    v16bf v;
    int4  q2[2];
    unsigned u[8];
};

// f32 -> bf16 round-to-nearest-even (used once per element in pack kernels)
__device__ __forceinline__ __bf16 f2bf(float f) {
    union { float f; unsigned u; } v; v.f = f;
    unsigned r = v.u + 0x7FFFu + ((v.u >> 16) & 1u);
    unsigned short hb = (unsigned short)(r >> 16);
    __bf16 out;
    __builtin_memcpy(&out, &hb, sizeof(out));
    return out;
}

// pack two f32 into a bf16x2 dword by truncation (hot softmax path)
__device__ __forceinline__ unsigned pack_bf16_trunc(float a, float b) {
    union { float f; unsigned u; } x, y; x.f = a; y.f = b;
    return (y.u & 0xFFFF0000u) | (x.u >> 16);
}

// K-index for element e (0..15) of a 16-bit A/B fragment, lane half hi (0/1).
__device__ __forceinline__ int kmap16(int e, int hi) {
    int j = e >> 1, p = e & 1;
    int base = (j < 4) ? (2 * j) : (16 + 2 * (j - 4));
    return base + hi * 8 + p;
}
// inverse: kk (0..31) -> (element e, half hi)
__device__ __forceinline__ void kinv16(int kk, int& e, int& hi) {
    hi = (kk >> 3) & 1;
    int kk2 = kk & ~8;
    int j = (kk2 < 8) ? (kk2 >> 1) : (4 + ((kk2 - 16) >> 1));
    e = 2 * j + (kk & 1);
}

// relative-position table index for token pair (m, n)
__device__ __forceinline__ int didx(int m, int n) {
    const int dy = (m >> 5) - (n >> 5);
    const int dx = (m & 31) - (n & 31);
    return (dy + (H_ - 1)) * (2 * W_ - 1) + (dx + (W_ - 1));
}

__device__ __forceinline__ v16bf load_frag(const __bf16* p) {
    FragU f;
    f.q2[0] = ((const int4*)p)[0];   // global_load_b128 / ds_load_b128
    f.q2[1] = ((const int4*)p)[1];
    return f.v;
}

__device__ __forceinline__ v8f wmma_bf16(const v16bf a, const v16bf b, const v8f c) {
    return __builtin_amdgcn_wmma_f32_16x16x32_bf16(false, a, false, b, (short)0, c,
                                                   false, false);
}

__device__ __forceinline__ void zero8(v8f& v) {
#pragma unroll
    for (int i = 0; i < 8; ++i) v[i] = 0.0f;
}

#if HAVE_TDM
typedef unsigned tdm_u32x4 __attribute__((ext_vector_type(4)));
typedef int      tdm_i32x8 __attribute__((ext_vector_type(8)));
typedef int      tdm_i32x4 __attribute__((ext_vector_type(4)));

// TDM: contiguous 1-D f32 tile, global -> LDS (descriptor per CDNA5 ISA)
// This toolchain's builtin arity: (u32x4, i32x8, i32x4, i32x4, i32x8, i32 cpol)
__device__ __forceinline__ void tdm_load_1d_f32(const float* gsrc, float* ldst,
                                                int nElems) {
    const unsigned lds_off = (unsigned)(uintptr_t)ldst;   // LDS generic ptr low bits
    const unsigned long long ga = (unsigned long long)(uintptr_t)gsrc;
    tdm_u32x4 g0;
    g0[0] = 1u;                                           // count=1 (valid user D#)
    g0[1] = lds_off;                                      // lds_addr
    g0[2] = (unsigned)(ga & 0xFFFFFFFFu);                 // global_addr[31:0]
    g0[3] = (unsigned)((ga >> 32) & 0x01FFFFFFu) | (2u << 30);  // addr[56:32], type=2
    tdm_i32x8 g1;
    g1[0] = (int)(2u << 16);                              // data_size=2 (4 bytes)
    g1[1] = (int)(((unsigned)nElems & 0xFFFFu) << 16);    // tensor_dim0[15:0]
    g1[2] = (int)((((unsigned)nElems >> 16) & 0xFFFFu) | (1u << 16)); // dim0 hi, dim1=1
    g1[3] = (int)(((unsigned)nElems & 0xFFFFu) << 16);    // tile_dim0
    g1[4] = 1;                                            // tile_dim1=1
    g1[5] = nElems;                                       // tensor_dim0_stride lo
    g1[6] = 0;
    g1[7] = 0;
    tdm_i32x4 z4; z4[0] = z4[1] = z4[2] = z4[3] = 0;
    tdm_i32x8 z8;
#pragma unroll
    for (int i = 0; i < 8; ++i) z8[i] = 0;
    __builtin_amdgcn_tensor_load_to_lds(g0, g1, z4, z4, z8, 0);
}
#endif

// =====================================================================
// delta table: delta[h][dy+31][dx+31] = rpb[idx,h] + 0.01*exp(-factor_h*dis)
// Collapses 67M per-element exp/gather work into 31,752 entries.
// =====================================================================
__global__ void __launch_bounds__(64)
build_delta_kernel(const float* __restrict__ rpb, const float* __restrict__ headsita,
                   float* __restrict__ delta) {
    const int h = blockIdx.y;
    const int i = blockIdx.x * 64 + threadIdx.x;
    if (i >= NDELTA) return;
    const int dy = i / 63 - 31, dx = i % 63 - 31;
    const float sita = headsita[h];
    const float factor = 1.0f / (2.0f * sita * sita + 1e-10f);
    const float dis = (float)(dy * dy + dx * dx) * (1.0f / 1024.0f);
    delta[h * NDELTA + i] = rpb[(size_t)i * HEADS + h] + 0.01f * __expf(-factor * dis);
}

// =====================================================================
// pack_frags: row-major f32 matrix -> bf16 WMMA fragment tiles.
// =====================================================================
__global__ void __launch_bounds__(32)
pack_frags_kernel(const float* __restrict__ src, int rowStride, int kTiles,
                  size_t zSrcStride, __bf16* __restrict__ dst, size_t zDstStride) {
    const int lane = threadIdx.x & 15, hi = threadIdx.x >> 4;
    const int rt = blockIdx.x, t = blockIdx.y, z = blockIdx.z;
    const float* row = src + (size_t)z * zSrcStride +
                       (size_t)(rt * 16 + lane) * rowStride + t * 32;
    FragU f;
#pragma unroll
    for (int e = 0; e < 16; ++e) f.v[e] = f2bf(row[kmap16(e, hi)]);
    __bf16* out = dst + (size_t)z * zDstStride +
                  ((size_t)rt * kTiles + t) * 512 + threadIdx.x * 16;
    ((int4*)out)[0] = f.q2[0];
    ((int4*)out)[1] = f.q2[1];
}

// =====================================================================
// pack_v: v as B-operand of attn@v (K-dim = token index).
// =====================================================================
__global__ void __launch_bounds__(32)
pack_v_kernel(const float* __restrict__ qkv_cat, __bf16* __restrict__ vbf) {
    const int lane = threadIdx.x & 15, hi = threadIdx.x >> 4;
    const int tc = blockIdx.x;          // 0..31
    const int bh = blockIdx.y;          // 0..63
    const float* vsrc = qkv_cat + (size_t)bh * N_TOK * (3 * DH) + 2 * DH;
#pragma unroll
    for (int g = 0; g < 4; ++g) {
        FragU f;
#pragma unroll
        for (int e = 0; e < 16; ++e)
            f.v[e] = f2bf(vsrc[(size_t)(tc * 32 + kmap16(e, hi)) * (3 * DH) +
                               g * 16 + lane]);
        __bf16* out = vbf + (((size_t)bh * 32 + tc) * 4 + g) * 512 + threadIdx.x * 16;
        ((int4*)out)[0] = f.q2[0];
        ((int4*)out)[1] = f.q2[1];
    }
}

// =====================================================================
// Kernel 1: qkv = x @ w_qkv^T, pure b128 + WMMA. Writes qkv_cat [B,h,N,3d].
// =====================================================================
__global__ void __launch_bounds__(32)
qkv_gemm_kernel(const __bf16* __restrict__ xbf, const __bf16* __restrict__ wbf,
                float* __restrict__ qkv_cat) {
    const int lane = threadIdx.x & 15, hi = threadIdx.x >> 4;
    const int rt = blockIdx.x;          // 0..511
    const int ct = blockIdx.y;          // 0..23

    v8f acc[4];
#pragma unroll
    for (int g = 0; g < 4; ++g) zero8(acc[g]);

    const __bf16* aBase = xbf + (size_t)rt * 16 * 512;

    for (int t = 0; t < DIM / 32; ++t) {
        if (t + 1 < DIM / 32)
            __builtin_prefetch(aBase + (size_t)(t + 1) * 512, 0, 0);
        const v16bf a = load_frag(aBase + (size_t)t * 512 + threadIdx.x * 16);
#pragma unroll
        for (int g = 0; g < 4; ++g) {
            const int c16 = ct * 4 + g;
            const v16bf b =
                load_frag(wbf + ((size_t)c16 * 16 + t) * 512 + threadIdx.x * 16);
            acc[g] = wmma_bf16(a, b, acc[g]);
        }
    }

#pragma unroll
    for (int g = 0; g < 4; ++g) {
        const int c    = ct * 64 + g * 16 + lane;
        const int part = c >> 9;
        const int rem  = c & 511;
        const int head = rem >> 6;
        const int d    = rem & 63;
#pragma unroll
        for (int r = 0; r < 8; ++r) {
            const int mloc = r + hi * 8;
            const int grow = rt * 16 + mloc;
            const int b    = grow >> 10;
            const int n    = grow & 1023;
            const size_t bhn = ((size_t)b * HEADS + head) * N_TOK + n;
            qkv_cat[bhn * (3 * DH) + part * DH + d] = acc[g][r];
        }
    }
}

// =====================================================================
// Kernel 2: fused attention per (b, h, 16-row tile).
// LDS: sD0 64KB + sums + delta 15.9KB + attn-frag 32KB  (~114 KB)
// =====================================================================
__global__ void __launch_bounds__(32)
attn_kernel(const __bf16* __restrict__ qbf, const __bf16* __restrict__ kbf,
            const __bf16* __restrict__ vbf, const float* __restrict__ deltaTbl,
            float* __restrict__ attn0_out, __bf16* __restrict__ ohbf) {
    extern __shared__ float smem[];
    float* sD0    = smem;                        // raw dots0 [16][1024] f32
    float* sSum   = smem + 16 * 1024;            // row sums [16]
    float* sDelta = sSum + 16;                   // 3969 (+3 pad) f32
    unsigned* sAttnU = (unsigned*)(sDelta + 3972);  // exp(attn) bf16 frags, 32 KB
    __bf16*   sAttnH = (__bf16*)sAttnU;

    const int lane = threadIdx.x & 15;
    const int hi   = threadIdx.x >> 4;
    const int mt   = blockIdx.x;                 // 0..63
    const int h    = blockIdx.y;
    const int b    = blockIdx.z;
    const int bh   = b * HEADS + h;
    const float scale = 0.125f;                  // DH^-0.5

    // kick off async delta-table load; overlapped with the whole q.k^T phase
#if HAVE_TDM
    tdm_load_1d_f32(deltaTbl + (size_t)h * NDELTA, sDelta, NDELTA);
#else
    for (int i = threadIdx.x; i < NDELTA; i += 32)
        sDelta[i] = deltaTbl[(size_t)h * NDELTA + i];
#endif

    // ---- Q fragments ----
    v16bf aq[2];
#pragma unroll
    for (int t = 0; t < 2; ++t)
        aq[t] = load_frag(qbf + (((size_t)bh * 64 + mt) * 2 + t) * 512 +
                          threadIdx.x * 16);

    // ---- phase 1: raw scores q.k^T * scale into LDS ----
    for (int j = 0; j < 64; ++j) {
        const __bf16* kt = kbf + (((size_t)bh * 64 + j) * 2) * 512;
        if (j + 1 < 64) __builtin_prefetch(kt + 1024, 0, 0);
        const v16bf bk0 = load_frag(kt + threadIdx.x * 16);
        const v16bf bk1 = load_frag(kt + 512 + threadIdx.x * 16);
        v8f c; zero8(c);
        c = wmma_bf16(aq[0], bk0, c);
        c = wmma_bf16(aq[1], bk1, c);
#pragma unroll
        for (int r = 0; r < 8; ++r) {
            const int mloc = r + hi * 8;
            sD0[mloc * 1024 + j * 16 + lane] = c[r] * scale;
        }
    }
#if HAVE_TDM
    __builtin_amdgcn_s_wait_tensorcnt(0);
#endif
    __syncthreads();

    // ---- phase 2: dual softmax (dots0 -> attn0 out; dots0+delta -> bf16 frags) ----
    const int tl = threadIdx.x;
    for (int r = 0; r < 16; ++r) {
        const int m = mt * 16 + r;
        float mx = -3.4e38f, mxd = -3.4e38f;
        for (int cI = tl; cI < 1024; cI += 32) {
            const float raw = sD0[r * 1024 + cI];
            mx  = fmaxf(mx, raw);
            mxd = fmaxf(mxd, raw + sDelta[didx(m, cI)]);
        }
#pragma unroll
        for (int s = 16; s > 0; s >>= 1) {
            mx  = fmaxf(mx,  __shfl_xor(mx,  s, 32));
            mxd = fmaxf(mxd, __shfl_xor(mxd, s, 32));
        }
        float sm = 0.0f, smd = 0.0f;
        for (int p = tl; p < 512; p += 32) {          // column pair (2p, 2p+1)
            const int col = 2 * p;
            const float raw0 = sD0[r * 1024 + col];
            const float raw1 = sD0[r * 1024 + col + 1];
            const float ev0 = __expf(raw0 - mx);
            const float ev1 = __expf(raw1 - mx);
            sm += ev0 + ev1;
            sD0[r * 1024 + col]     = ev0;            // overwrite raw with exp
            sD0[r * 1024 + col + 1] = ev1;
            const float ed0 = __expf(raw0 + sDelta[didx(m, col)] - mxd);
            const float ed1 = __expf(raw1 + sDelta[didx(m, col + 1)] - mxd);
            smd += ed0 + ed1;
            const int tt = col >> 5, kk = col & 31;
            int e2, hi2; kinv16(kk, e2, hi2);
            sAttnU[tt * 256 + (r + 16 * hi2) * 8 + (e2 >> 1)] =
                pack_bf16_trunc(ed0, ed1);
        }
#pragma unroll
        for (int s = 16; s > 0; s >>= 1) {
            sm  += __shfl_xor(sm,  s, 32);
            smd += __shfl_xor(smd, s, 32);
        }
        const float inv = 1.0f / sm;
        float* a0row = attn0_out + (((size_t)bh) * N_TOK + m) * N_TOK;
        for (int cI = tl; cI < 1024; cI += 32)
            a0row[cI] = sD0[r * 1024 + cI] * inv;
        if (tl == 0) sSum[r] = smd;
    }
    __syncthreads();

    // ---- phase 3: out = (exp-attn @ v) / rowsum; A from LDS, B from vbf ----
    v8f acc[4];
#pragma unroll
    for (int g = 0; g < 4; ++g) zero8(acc[g]);
    for (int t = 0; t < N_TOK / 32; ++t) {
        const v16bf a = load_frag(sAttnH + (size_t)t * 512 + threadIdx.x * 16);
#pragma unroll
        for (int g = 0; g < 4; ++g) {
            const v16bf bv = load_frag(vbf + (((size_t)bh * 32 + t) * 4 + g) * 512 +
                                       threadIdx.x * 16);
            acc[g] = wmma_bf16(a, bv, acc[g]);
        }
    }

    // scatter-store OH as bf16 fragments for the out-projection A operand
#pragma unroll
    for (int g = 0; g < 4; ++g) {
        const int d = g * 16 + lane;
        const int k = h * 64 + d;
        const int tt = k >> 5, kk = k & 31;
        int e2, hi2; kinv16(kk, e2, hi2);
#pragma unroll
        for (int r = 0; r < 8; ++r) {
            const int mloc = r + hi * 8;
            const float val = acc[g][r] / sSum[mloc];
            const size_t rt = (size_t)b * 64 + mt;
            ohbf[(rt * 16 + tt) * 512 + (mloc + 16 * hi2) * 16 + e2] = f2bf(val);
        }
    }
}

// =====================================================================
// Kernel 3: out = OH @ w_out^T + b_out, pure b128 + WMMA.
// =====================================================================
__global__ void __launch_bounds__(32)
out_proj_kernel(const __bf16* __restrict__ ohbf, const __bf16* __restrict__ wbf,
                const float* __restrict__ bout, float* __restrict__ out) {
    const int lane = threadIdx.x & 15, hi = threadIdx.x >> 4;
    const int rt   = blockIdx.x;        // 0..511
    const int ct   = blockIdx.y;        // 0..7

    v8f acc[4];
#pragma unroll
    for (int g = 0; g < 4; ++g) zero8(acc[g]);

    const __bf16* aBase = ohbf + (size_t)rt * 16 * 512;
    for (int t = 0; t < INNER / 32; ++t) {
        if (t + 1 < INNER / 32)
            __builtin_prefetch(aBase + (size_t)(t + 1) * 512, 0, 0);
        const v16bf a = load_frag(aBase + (size_t)t * 512 + threadIdx.x * 16);
#pragma unroll
        for (int g = 0; g < 4; ++g) {
            const int c16 = ct * 4 + g;
            const v16bf b =
                load_frag(wbf + ((size_t)c16 * 16 + t) * 512 + threadIdx.x * 16);
            acc[g] = wmma_bf16(a, b, acc[g]);
        }
    }
#pragma unroll
    for (int g = 0; g < 4; ++g)
#pragma unroll
        for (int r = 0; r < 8; ++r) {
            const int mloc = r + hi * 8;
            const int c = ct * 64 + g * 16 + lane;
            const int growm = rt * 16 + mloc;
            out[(size_t)growm * DIM + c] = acc[g][r] + bout[c];
        }
}

// =====================================================================
extern "C" void kernel_launch(void* const* d_in, const int* in_sizes, int n_in,
                              void* d_out, int out_size, void* d_ws, size_t ws_size,
                              hipStream_t stream) {
    (void)in_sizes; (void)n_in; (void)out_size; (void)ws_size;

    const float* x    = (const float*)d_in[0];
    const float* wqkv = (const float*)d_in[1];
    const float* wout = (const float*)d_in[2];
    const float* bout = (const float*)d_in[3];
    const float* sita = (const float*)d_in[4];
    const float* rpb  = (const float*)d_in[5];

    // outputs concatenated flat: out, qkv_cat, attn0
    float* out_proj = (float*)d_out;
    float* qkv_cat  = out_proj + (size_t)B_ * N_TOK * DIM;
    float* attn0    = qkv_cat + (size_t)B_ * HEADS * N_TOK * 3 * DH;

    // workspaces
    char* p = (char*)d_ws;
    __bf16* xbf    = (__bf16*)p; p += (size_t)8192 * 512 * 2;
    __bf16* wqkvbf = (__bf16*)p; p += (size_t)1536 * 512 * 2;
    __bf16* woutbf = (__bf16*)p; p += (size_t)512 * 512 * 2;
    __bf16* qbf    = (__bf16*)p; p += (size_t)64 * 1024 * 64 * 2;
    __bf16* kbf    = (__bf16*)p; p += (size_t)64 * 1024 * 64 * 2;
    __bf16* vbf    = (__bf16*)p; p += (size_t)64 * 1024 * 64 * 2;
    __bf16* ohbf   = (__bf16*)p; p += (size_t)8192 * 512 * 2;
    float*  deltaT = (float*)p;  p += (size_t)HEADS * NDELTA * 4;

    build_delta_kernel<<<dim3((NDELTA + 63) / 64, HEADS), 64, 0, stream>>>(
        rpb, sita, deltaT);

    pack_frags_kernel<<<dim3(512, 16, 1), 32, 0, stream>>>(x, DIM, 16, 0, xbf, 0);
    pack_frags_kernel<<<dim3(96, 16, 1), 32, 0, stream>>>(wqkv, DIM, 16, 0, wqkvbf, 0);
    pack_frags_kernel<<<dim3(32, 16, 1), 32, 0, stream>>>(wout, INNER, 16, 0, woutbf, 0);

    qkv_gemm_kernel<<<dim3(512, 24), 32, 0, stream>>>(xbf, wqkvbf, qkv_cat);

    const size_t zSrc = (size_t)N_TOK * 3 * DH;
    const size_t zDst = (size_t)64 * 2 * 512;
    pack_frags_kernel<<<dim3(64, 2, 64), 32, 0, stream>>>(qkv_cat, 3 * DH, 2, zSrc,
                                                          qbf, zDst);
    pack_frags_kernel<<<dim3(64, 2, 64), 32, 0, stream>>>(qkv_cat + DH, 3 * DH, 2,
                                                          zSrc, kbf, zDst);
    pack_v_kernel<<<dim3(32, 64), 32, 0, stream>>>(qkv_cat, vbf);

    // LDS: sD0 64KB + sums/delta ~16KB + attn bf16 frags 32KB
    const size_t smem_bytes = (16 * 1024 + 16 + 3972) * sizeof(float) + 16 * 1024 * 2;
    attn_kernel<<<dim3(64, HEADS, B_), 32, smem_bytes, stream>>>(
        qbf, kbf, vbf, deltaT, attn0, ohbf);

    out_proj_kernel<<<dim3(512, 8), 32, 0, stream>>>(ohbf, woutbf, bout, out_proj);
}
